// GNN_87316685128359
// MI455X (gfx1250) — matrix operand order repaired and checked by reading, not compiled, not run on hardware
//
#include <hip/hip_runtime.h>

typedef __attribute__((ext_vector_type(16))) __bf16          v16bf;
typedef __attribute__((ext_vector_type(16))) unsigned short  v16us;
typedef __attribute__((ext_vector_type(8)))  float           v8f;

#define HID    128
#define NGRAPH 64
#define DENSE  256
#define OUTD   8

__device__ __forceinline__ unsigned short f32_to_bf16(float x) {
  unsigned u = __builtin_bit_cast(unsigned, x);
  u += 0x7FFFu + ((u >> 16) & 1u);          // round-to-nearest-even (approx for ties)
  return (unsigned short)(u >> 16);
}
__device__ __forceinline__ float bf16u_to_f32(unsigned short h) {
  unsigned u = ((unsigned)h) << 16;
  return __builtin_bit_cast(float, u);
}

// ---------------------------------------------------------------------------
// GEMM + fused GCN self-loop epilogue.
//   H[N x 128]   = X[N x 128] @ W[128 x 128]            (fp32 via 4x bf16 WMMA)
//   AGG[N x 128] = dis[v]^2 * H + bias                  (aggregation init)
// X and W are staged in LDS pre-swizzled into the exact WMMA A/B fragment
// layouts (hi/lo bf16 planes), so each fragment is ONE contiguous 32B vector
// load per plane (ds_load_b128 x2) instead of 16 scalar u16 loads + packing.
//   A-frag layout index: [wave(8)][kchunk(4)][lane(32)][j(16)]
//   B-frag layout index: [kchunk(4)][tile(8)][lane(32)][j(16)]
// with (this code's contiguous k order, shared by A and B):
//   A: M = lane&15,  K = ck*32 + (lane&16?16:0) + j
//   B: N = lane&15,  K = ck*32 + (lane&16?16:0) + j
// ---------------------------------------------------------------------------
__global__ __launch_bounds__(256)
void k_gemm128_bf16x4(const float* __restrict__ X, const float* __restrict__ W,
                      const float* __restrict__ dis, const float* __restrict__ bias,
                      float* __restrict__ H, float* __restrict__ AGG, int N)
{
  extern __shared__ char smem[];
  unsigned short* Wh = (unsigned short*)smem;   // 16384 bf16 = 32KB
  unsigned short* Wl = Wh + 16384;              // 32KB
  unsigned short* Xh = Wl + 16384;              // 32KB
  unsigned short* Xl = Xh + 16384;              // 32KB   (total 128KB dynamic LDS)

  const int tid = threadIdx.x;
  const int block_row = blockIdx.x * 128;

  // Stage W into B-fragment order (hi/lo bf16)
  for (int i = tid; i < 16384; i += 256) {
    int j  = i & 15;
    int L  = (i >> 4) & 31;
    int t  = (i >> 9) & 7;
    int ck = i >> 12;
    int K  = ck*32 + ((L & 16) ? 16 : 0) + j;
    int n  = t*16 + (L & 15);
    float w = W[K*128 + n];
    unsigned short hb = f32_to_bf16(w);
    Wh[i] = hb;
    Wl[i] = f32_to_bf16(w - bf16u_to_f32(hb));
  }
  // Stage X tile into A-fragment order (hi/lo bf16); zero-fill OOB rows so
  // EXEC stays all-ones for the WMMAs.
  for (int i = tid; i < 16384; i += 256) {
    int j  = i & 15;
    int L  = (i >> 4) & 31;
    int ck = (i >> 9) & 3;
    int wv = i >> 11;
    int K  = ck*32 + ((L & 16) ? 16 : 0) + j;
    int row = block_row + wv*16 + (L & 15);
    float a = (row < N) ? X[(size_t)row*128 + K] : 0.0f;
    unsigned short hb = f32_to_bf16(a);
    Xh[i] = hb;
    Xl[i] = f32_to_bf16(a - bf16u_to_f32(hb));
  }
  __syncthreads();

  const int lane = tid & 31;
  const int wave = tid >> 5;

  v8f acc[8];
#pragma unroll
  for (int t = 0; t < 8; ++t) acc[t] = v8f{0.f,0.f,0.f,0.f,0.f,0.f,0.f,0.f};

#pragma unroll
  for (int ck = 0; ck < 4; ++ck) {
    const int abase = ((wave*4 + ck)*32 + lane) * 16;
    v16bf ahi = __builtin_bit_cast(v16bf, *(const v16us*)(Xh + abase));
    v16bf alo = __builtin_bit_cast(v16bf, *(const v16us*)(Xl + abase));
#pragma unroll
    for (int t = 0; t < 8; ++t) {
      const int bbase = ((ck*8 + t)*32 + lane) * 16;
      v16bf bhi = __builtin_bit_cast(v16bf, *(const v16us*)(Wh + bbase));
      v16bf blo = __builtin_bit_cast(v16bf, *(const v16us*)(Wl + bbase));
      acc[t] = __builtin_amdgcn_wmma_f32_16x16x32_bf16(false, ahi, false, bhi, (short)0, acc[t], false, false);
      acc[t] = __builtin_amdgcn_wmma_f32_16x16x32_bf16(false, ahi, false, blo, (short)0, acc[t], false, false);
      acc[t] = __builtin_amdgcn_wmma_f32_16x16x32_bf16(false, alo, false, bhi, (short)0, acc[t], false, false);
      acc[t] = __builtin_amdgcn_wmma_f32_16x16x32_bf16(false, alo, false, blo, (short)0, acc[t], false, false);
    }
  }

  // C/D layout: VGPR r -> row = r + (lane>=16?8:0), col = lane&15.
  // Fused epilogue: H = acc,  AGG = dis[row]^2 * acc + bias[col].
  const int half8 = (lane & 16) ? 8 : 0;
#pragma unroll
  for (int r = 0; r < 8; ++r) {
    int row = block_row + wave*16 + r + half8;
    if (row < N) {
      float dv = dis[row];
      float d2 = dv * dv;
#pragma unroll
      for (int t = 0; t < 8; ++t) {
        int col = t*16 + (lane & 15);
        float v = acc[t][r];
        H  [(size_t)row*128 + col] = v;
        AGG[(size_t)row*128 + col] = d2 * v + bias[col];
      }
    }
  }
}

// ---------------------------------------------------------------------------
// Degree / normalization
// ---------------------------------------------------------------------------
__global__ void k_deg_init(float* deg, int n) {
  int i = blockIdx.x*256 + threadIdx.x;
  if (i < n) deg[i] = 1.0f;                      // self-loop
}
__global__ void k_deg_edges(const int* __restrict__ dst, float* deg, int e) {
  int i = blockIdx.x*256 + threadIdx.x;
  if (i < e) atomicAdd(&deg[dst[i]], 1.0f);
}
__global__ void k_deg_rsqrt(float* deg, int n) {
  int i = blockIdx.x*256 + threadIdx.x;
  if (i < n) deg[i] = rsqrtf(deg[i]);
}

// ---------------------------------------------------------------------------
// GCN edge aggregation: AGG[d] += dis[s]*dis[d] * H[s]   (one wave per edge)
// ---------------------------------------------------------------------------
__global__ void k_agg_edges(const int* __restrict__ src, const int* __restrict__ dst,
                            const float* __restrict__ dis, const float* __restrict__ h,
                            float* __restrict__ out, int e) {
  long long tid = (long long)blockIdx.x*256 + threadIdx.x;
  int eidx = (int)(tid >> 5);
  if (eidx >= e) return;
  int c = (int)(tid & 31);                       // lane -> float4 chunk
  int s = src[eidx], d = dst[eidx];
  float w = dis[s] * dis[d];
  float4 x = ((const float4*)(h + (size_t)s*HID))[c];
  float* o = out + (size_t)d*HID + c*4;
  atomicAdd(o+0, w*x.x); atomicAdd(o+1, w*x.y);
  atomicAdd(o+2, w*x.z); atomicAdd(o+3, w*x.w);
}

// ---------------------------------------------------------------------------
// LayerNorm(ReLU(x)) — one wave32 per node, 4 features/lane
// ---------------------------------------------------------------------------
__global__ __launch_bounds__(256)
void k_relu_ln(const float* __restrict__ in, float* __restrict__ out,
               const float* __restrict__ g, const float* __restrict__ b, int n) {
  int wave = threadIdx.x >> 5, lane = threadIdx.x & 31;
  int v = blockIdx.x*8 + wave;
  if (v >= n) return;
  float4 x = ((const float4*)(in + (size_t)v*HID))[lane];
  x.x = fmaxf(x.x, 0.f); x.y = fmaxf(x.y, 0.f);
  x.z = fmaxf(x.z, 0.f); x.w = fmaxf(x.w, 0.f);
  float s = x.x + x.y + x.z + x.w;
#pragma unroll
  for (int m = 16; m >= 1; m >>= 1) s += __shfl_xor(s, m, 32);
  float mu = s * (1.0f/128.0f);
  float d0 = x.x-mu, d1 = x.y-mu, d2 = x.z-mu, d3 = x.w-mu;
  float q = d0*d0 + d1*d1 + d2*d2 + d3*d3;
#pragma unroll
  for (int m = 16; m >= 1; m >>= 1) q += __shfl_xor(q, m, 32);
  float inv = rsqrtf(q * (1.0f/128.0f) + 1e-5f);
  float4 gg = ((const float4*)g)[lane];
  float4 bb = ((const float4*)b)[lane];
  float4 y;
  y.x = d0*inv*gg.x + bb.x; y.y = d1*inv*gg.y + bb.y;
  y.z = d2*inv*gg.z + bb.z; y.w = d3*inv*gg.w + bb.w;
  ((float4*)(out + (size_t)v*HID))[lane] = y;
}

// ---------------------------------------------------------------------------
// Pooling + MLP head + log-softmax
// ---------------------------------------------------------------------------
__global__ void k_zero(float* p, int n) {
  int i = blockIdx.x*256 + threadIdx.x;
  if (i < n) p[i] = 0.0f;
}
__global__ void k_pool_sum(const float* __restrict__ h, const int* __restrict__ batch,
                           float* sums, int n) {
  int idx = blockIdx.x*256 + threadIdx.x;
  if (idx < n*HID) {
    int v = idx >> 7, f = idx & 127;
    atomicAdd(&sums[batch[v]*HID + f], h[idx]);
  }
}
__global__ void k_pool_cnt(const int* __restrict__ batch, float* cnt, int n) {
  int i = blockIdx.x*256 + threadIdx.x;
  if (i < n) atomicAdd(&cnt[batch[i]], 1.0f);
}
__global__ void k_pool_div(const float* __restrict__ sums, const float* __restrict__ cnt,
                           float* pooled) {
  int idx = blockIdx.x*256 + threadIdx.x;
  if (idx < NGRAPH*HID) {
    int gidx = idx >> 7;
    pooled[idx] = sums[idx] / fmaxf(cnt[gidx], 1.0f);
  }
}
__global__ void k_dense(const float* __restrict__ X, const float* __restrict__ W,
                        const float* __restrict__ b, float* __restrict__ Y,
                        int R, int Cin, int Cout) {
  int idx = blockIdx.x*256 + threadIdx.x;
  if (idx >= R*Cout) return;
  int r = idx / Cout, c = idx % Cout;
  float acc = b[c];
  for (int k = 0; k < Cin; ++k) acc += X[r*Cin + k] * W[k*Cout + c];
  Y[idx] = acc;
}
__global__ void k_logsoftmax(const float* __restrict__ logits,
                             float* __restrict__ out_logits, float* __restrict__ out_probs) {
  int t = threadIdx.x;
  if (t >= NGRAPH) return;
  float l[OUTD]; float m = -1e30f;
#pragma unroll
  for (int i = 0; i < OUTD; ++i) { l[i] = logits[t*OUTD + i]; m = fmaxf(m, l[i]); }
  float s = 0.f;
#pragma unroll
  for (int i = 0; i < OUTD; ++i) s += __expf(l[i] - m);
  float lse = __logf(s);
#pragma unroll
  for (int i = 0; i < OUTD; ++i) {
    out_logits[t*OUTD + i] = l[i];
    out_probs [t*OUTD + i] = l[i] - m - lse;
  }
}

// ---------------------------------------------------------------------------
extern "C" void kernel_launch(void* const* d_in, const int* in_sizes, int n_in,
                              void* d_out, int out_size, void* d_ws, size_t ws_size,
                              hipStream_t stream) {
  (void)n_in; (void)out_size; (void)ws_size;
  const float* x     = (const float*)d_in[0];
  const int*   ei    = (const int*)  d_in[1];
  const int*   batch = (const int*)  d_in[2];
  const float* W1    = (const float*)d_in[3];
  const float* b1    = (const float*)d_in[4];
  const float* ln1_g = (const float*)d_in[5];
  const float* ln1_b = (const float*)d_in[6];
  const float* Wc    = (const float*)d_in[7];
  const float* bc    = (const float*)d_in[8];
  const float* ln_g  = (const float*)d_in[9];
  const float* ln_b  = (const float*)d_in[10];
  const float* Wm1   = (const float*)d_in[11];
  const float* bm1   = (const float*)d_in[12];
  const float* Wm2   = (const float*)d_in[13];
  const float* bm2   = (const float*)d_in[14];
  const float* Wm3   = (const float*)d_in[15];
  const float* bm3   = (const float*)d_in[16];

  const int N = in_sizes[2];          // N_NODES (batch vector length)
  const int E = in_sizes[1] / 2;      // edge_index is [2, E]
  const int* src = ei;
  const int* dst = ei + E;

  float* ws = (float*)d_ws;
  float* dis    = ws;  ws += N;
  float* bufA   = ws;  ws += (size_t)N*HID;   // GEMM output h = in @ W
  float* bufC   = ws;  ws += (size_t)N*HID;   // aggregated / normalized features
  float* sums   = ws;  ws += NGRAPH*HID;
  float* cnt    = ws;  ws += NGRAPH;          // contiguous with sums (zeroed together)
  float* pooled = ws;  ws += NGRAPH*HID;
  float* z1     = ws;  ws += NGRAPH*DENSE;
  float* z2     = ws;  ws += NGRAPH*(DENSE/2);
  float* lgts   = ws;  ws += NGRAPH*OUTD;

  float* emb        = (float*)d_out;
  float* out_logits = emb + (size_t)N*HID;
  float* out_probs  = out_logits + NGRAPH*OUTD;

  auto cdiv = [](long long a, long long b) { return (int)((a + b - 1) / b); };

  // symmetric GCN normalization coefficients
  k_deg_init <<<cdiv(N,256),256,0,stream>>>(dis, N);
  k_deg_edges<<<cdiv(E,256),256,0,stream>>>(dst, dis, E);
  k_deg_rsqrt<<<cdiv(N,256),256,0,stream>>>(dis, N);

  const int    gemm_blocks = cdiv(N, 128);
  const size_t gemm_lds    = 4 * 16384 * 2;   // 128 KB (<320 KB WGP LDS)
  const long long NH  = (long long)N * HID;
  const long long E32 = (long long)E * 32;

  for (int layer = 0; layer < 3; ++layer) {
    const float* gin  = (layer == 0) ? x  : bufC;
    const float* W    = (layer == 0) ? W1 : Wc + (size_t)(layer-1)*HID*HID;
    const float* bias = (layer == 0) ? b1 : bc + (size_t)(layer-1)*HID;
    const float* lg_  = (layer == 0) ? ln1_g : ln_g + (size_t)(layer-1)*HID;
    const float* lb_  = (layer == 0) ? ln1_b : ln_b + (size_t)(layer-1)*HID;
    float* agg = (layer == 2) ? emb : bufC;   // last conv output IS `emb` in d_out

    // GEMM (in-place safe: each block reads only its own 128 rows of gin
    // during staging, then writes those same rows of agg in the epilogue)
    k_gemm128_bf16x4<<<gemm_blocks,256,gemm_lds,stream>>>(gin, W, dis, bias, bufA, agg, N);
    k_agg_edges<<<cdiv(E32,256),256,0,stream>>>(src, dst, dis, bufA, agg, E);
    k_relu_ln  <<<cdiv(N,8),    256,0,stream>>>(agg, bufC, lg_, lb_, N);
  }

  // global mean pool + MLP head
  k_zero    <<<cdiv(NGRAPH*HID + NGRAPH,256),256,0,stream>>>(sums, NGRAPH*HID + NGRAPH);
  k_pool_sum<<<cdiv(NH,256),256,0,stream>>>(bufC, batch, sums, N);
  k_pool_cnt<<<cdiv(N,256), 256,0,stream>>>(batch, cnt, N);
  k_pool_div<<<cdiv(NGRAPH*HID,256),256,0,stream>>>(sums, cnt, pooled);
  k_dense<<<cdiv(NGRAPH*DENSE,256),    256,0,stream>>>(pooled, Wm1, bm1, z1, NGRAPH, HID,     DENSE);
  k_dense<<<cdiv(NGRAPH*(DENSE/2),256),256,0,stream>>>(z1,     Wm2, bm2, z2, NGRAPH, DENSE,   DENSE/2);
  k_dense<<<cdiv(NGRAPH*OUTD,256),     256,0,stream>>>(z2,     Wm3, bm3, lgts, NGRAPH, DENSE/2, OUTD);
  k_logsoftmax<<<1,64,0,stream>>>(lgts, out_logits, out_probs);
}
